// EnhancedTransformerBlock_80169859548047
// MI455X (gfx1250) — matrix operand order repaired and checked by reading, compiled, not run on hardware
//
#include <hip/hip_runtime.h>

// ---------------------------------------------------------------------------
// Types for CDNA5 WMMA (wave32): v_wmma_f32_16x16x32_bf16
// ---------------------------------------------------------------------------
typedef __attribute__((ext_vector_type(16))) __bf16 v16bf;
typedef __attribute__((ext_vector_type(8)))  float  v8f;
typedef __attribute__((ext_vector_type(4)))  int    v4i;

union FragU   { uint4 u[2]; v16bf v; };     // 32B fragment (16 bf16)
union HalfPack{ uint4 u; __bf16 h[8]; };    // 16B of bf16

typedef __attribute__((address_space(1))) v4i g_v4i;  // global int4
typedef __attribute__((address_space(3))) v4i l_v4i;  // LDS int4

#define EPI_BIAS     1
#define EPI_RESID    2
#define EPI_ROWSCALE 4
#define EPI_ACCUM    8
#define EPI_GELU     16

#if __has_builtin(__builtin_amdgcn_global_load_async_to_lds_b128)
#define HAVE_ASYNC_LDS 1
#else
#define HAVE_ASYNC_LDS 0
#endif

__device__ __forceinline__ void async_wait_le4() {
#if __has_builtin(__builtin_amdgcn_s_wait_asynccnt)
    __builtin_amdgcn_s_wait_asynccnt(4);
#else
    asm volatile("s_wait_asynccnt 0x4" ::: "memory");
#endif
}
__device__ __forceinline__ void async_wait_0() {
#if __has_builtin(__builtin_amdgcn_s_wait_asynccnt)
    __builtin_amdgcn_s_wait_asynccnt(0);
#else
    asm volatile("s_wait_asynccnt 0x0" ::: "memory");
#endif
}

// ---------------------------------------------------------------------------
// Weight convert f32[K][N] -> bf16 transposed [N][K]
// ---------------------------------------------------------------------------
__global__ __launch_bounds__(256) void cvt_transpose_kernel(const float* __restrict__ in,
                                                            __bf16* __restrict__ out,
                                                            int K, int N) {
    size_t i = (size_t)blockIdx.x * 256 + threadIdx.x;
    size_t total = (size_t)K * N;
    if (i < total) {
        int k = (int)(i / N), n = (int)(i % N);
        out[(size_t)n * K + k] = (__bf16)in[i];
    }
}

// ---------------------------------------------------------------------------
// LayerNorm over H=1024, bf16 output. One block (256 thr) per token.
// ---------------------------------------------------------------------------
__global__ __launch_bounds__(256) void ln_bf16_kernel(const float* __restrict__ x,
                                                      const float* __restrict__ g,
                                                      const float* __restrict__ b,
                                                      __bf16* __restrict__ out) {
    __shared__ float red[256];
    int t = threadIdx.x;
    size_t tok = blockIdx.x;
    const float* xr = x + tok * 1024;
    float v[4], s = 0.f;
#pragma unroll
    for (int i = 0; i < 4; ++i) { v[i] = xr[t + i * 256]; s += v[i]; }
    red[t] = s; __syncthreads();
    for (int o = 128; o > 0; o >>= 1) { if (t < o) red[t] += red[t + o]; __syncthreads(); }
    float mu = red[0] * (1.f / 1024.f);
    __syncthreads();
    float s2 = 0.f;
#pragma unroll
    for (int i = 0; i < 4; ++i) { float d = v[i] - mu; s2 += d * d; }
    red[t] = s2; __syncthreads();
    for (int o = 128; o > 0; o >>= 1) { if (t < o) red[t] += red[t + o]; __syncthreads(); }
    float inv = rsqrtf(red[0] * (1.f / 1024.f) + 1e-12f);
#pragma unroll
    for (int i = 0; i < 4; ++i) {
        int c = t + i * 256;
        out[tok * 1024 + c] = (__bf16)((v[i] - mu) * inv * g[c] + b[c]);
    }
}

// ---------------------------------------------------------------------------
// Generic bf16 GEMM with pre-transposed B (BT = [N][K] row-major, weights
// resident in 192MB L2). Block tile 128x128, K-step 64, 8 waves, wave tile
// 64x32 (4x2 WMMA). A tile double-buffered in LDS, staged with
// global_load_async_to_lds_b128 (ASYNCcnt) when available; B fragments are
// two contiguous global b128 loads per lane (no LDS for B at all).
// ---------------------------------------------------------------------------
__global__ __launch_bounds__(256) void gemm_bf16_kernel(
    const __bf16* __restrict__ A, int lda,
    const __bf16* __restrict__ BT, int ldbt,
    float* __restrict__ Cf, __bf16* __restrict__ Cb, int ldc,
    const float* __restrict__ bias,
    const float* __restrict__ resid,
    const float* __restrict__ rowscale, int rss,
    int M, int N, int K, int flags)
{
    __shared__ __align__(16) __bf16 sA[2][128 * 64];   // 2 x 16KB

    int tid = threadIdx.x;
    int lane = tid & 31, wave = tid >> 5;
    int hi = lane >> 4, lo = lane & 15;
    int wm = wave & 1, wn = wave >> 1;           // wave tile: rows wm*64, cols wn*32
    int bm = blockIdx.y * 128, bn = blockIdx.x * 128;

    v8f acc[4][2] = {};

    // A staging: thread -> 32 contiguous halves (64B) of one row
    int arow = tid >> 1, aseg = (tid & 1) * 32;
    const __bf16* agbase = A + (size_t)(bm + arow) * lda + aseg;
    int aldsoff = arow * 64 + aseg;

    auto stage = [&](int buf, int k0) {
        const __bf16* g = agbase + k0;
        __bf16* l = &sA[buf][aldsoff];
#if HAVE_ASYNC_LDS
        __builtin_amdgcn_global_load_async_to_lds_b128((g_v4i*)g, (l_v4i*)l, 0, 0);
        __builtin_amdgcn_global_load_async_to_lds_b128((g_v4i*)g, (l_v4i*)l, 16, 0);
        __builtin_amdgcn_global_load_async_to_lds_b128((g_v4i*)g, (l_v4i*)l, 32, 0);
        __builtin_amdgcn_global_load_async_to_lds_b128((g_v4i*)g, (l_v4i*)l, 48, 0);
#else
        const uint4* gu = (const uint4*)g;
        uint4* lu = (uint4*)l;
        lu[0] = gu[0]; lu[1] = gu[1]; lu[2] = gu[2]; lu[3] = gu[3];
#endif
    };

    int nst = K / 64;
    int p = 0;
    stage(0, 0);
    for (int it = 0; it < nst; ++it) {
        int k0 = it * 64;
        __syncthreads();                       // all reads of buf p^1 (prev iter) done
        bool hasNext = (it + 1) < nst;
        if (hasNext) stage(p ^ 1, k0 + 64);    // overlap next tile with compute
        if (hasNext) async_wait_le4(); else async_wait_0();
        __syncthreads();                       // buf p fully resident for all waves

        const __bf16* sAb = &sA[p][0];
#pragma unroll
        for (int ks = 0; ks < 2; ++ks) {
            v16bf af[4], bfr[2];
#pragma unroll
            for (int mi = 0; mi < 4; ++mi) {
                // A frag: M = lo, 16 contiguous K at 16*hi within this k-step
                int base = (wm * 64 + mi * 16 + lo) * 64 + ks * 32 + hi * 16;
                FragU f;
                f.u[0] = *(const uint4*)(sAb + base);
                f.u[1] = *(const uint4*)(sAb + base + 8);
                af[mi] = f.v;
            }
#pragma unroll
            for (int ni = 0; ni < 2; ++ni) {
                // B frag: N = lo, 16 contiguous K at 16*hi — straight from global BT
                const __bf16* bp = BT + (size_t)(bn + wn * 32 + ni * 16 + lo) * ldbt
                                      + k0 + ks * 32 + hi * 16;
                FragU f;
                f.u[0] = *(const uint4*)bp;
                f.u[1] = *(const uint4*)(bp + 8);
                bfr[ni] = f.v;
            }
#pragma unroll
            for (int mi = 0; mi < 4; ++mi)
#pragma unroll
                for (int ni = 0; ni < 2; ++ni)
                    acc[mi][ni] = __builtin_amdgcn_wmma_f32_16x16x32_bf16(
                        false, af[mi], false, bfr[ni], (short)0, acc[mi][ni], false, false);
        }
        p ^= 1;
    }

    // Epilogue: C/D layout: row = r + 8*hi, col = lo
#pragma unroll
    for (int mi = 0; mi < 4; ++mi)
#pragma unroll
        for (int ni = 0; ni < 2; ++ni) {
            int col = bn + wn * 32 + ni * 16 + lo;
            if (col >= N) continue;
            float bv = (flags & EPI_BIAS) ? bias[col] : 0.f;
#pragma unroll
            for (int r = 0; r < 8; ++r) {
                int row = bm + wm * 64 + mi * 16 + r + 8 * hi;
                if (row >= M) continue;
                float v = acc[mi][ni][r] + bv;
                if (flags & EPI_GELU)     v = 0.5f * v * (1.f + erff(v * 0.70710678118f));
                if (flags & EPI_ROWSCALE) v *= rowscale[(size_t)row * rss];
                if (flags & EPI_RESID)    v += resid[(size_t)row * ldc + col];
                size_t o = (size_t)row * ldc + col;
                if (flags & EPI_ACCUM)    v += Cf[o];
                if (Cf) Cf[o] = v;
                if (Cb) Cb[o] = (__bf16)v;
            }
        }
}

// ---------------------------------------------------------------------------
// Causal flash attention. qkv bf16 [8192][3072] (q|k|v each [nh=16][hd=64]).
// grid (qtile=16, head=16, batch=4), block 256 (8 waves, 16 q-rows each).
// ---------------------------------------------------------------------------
__global__ __launch_bounds__(256) void attn_kernel(const __bf16* __restrict__ qkv,
                                                   __bf16* __restrict__ attn)
{
    __shared__ __align__(16) __bf16 sVT[64 * 32];       // V^T: [d][kv] for a 32-kv tile
    __shared__ __align__(16) __bf16 pbuf[8][16 * 32];   // per-wave P tile

    int tid = threadIdx.x, lane = tid & 31, w = tid >> 5;
    int hi = lane >> 4, lo = lane & 15;
    int qb = blockIdx.x * 128;
    int h  = blockIdx.y;
    int b  = blockIdx.z;
    const size_t ts = 3072;
    const __bf16* qbase = qkv + (size_t)b * 2048 * ts + h * 64;
    const __bf16* kbase = qbase + 1024;
    const __bf16* vbase = qbase + 2048;

    // Q A-fragments (2 K-steps over hd=64), kept in registers
    v16bf qa[2];
    int qrow = qb + w * 16 + lo;
#pragma unroll
    for (int ks = 0; ks < 2; ++ks) {
        const __bf16* p = qbase + (size_t)qrow * ts + ks * 32 + hi * 16;
        FragU f; f.u[0] = *(const uint4*)p; f.u[1] = *(const uint4*)(p + 8);
        qa[ks] = f.v;
    }

    v8f o[4] = {};
    float m[8], l[8];
#pragma unroll
    for (int r = 0; r < 8; ++r) { m[r] = -1e30f; l[r] = 0.f; }

    __bf16* pb = &pbuf[w][0];
    int npt = (qb + 128) / 32;                 // causal: kv < qb+128
    for (int t = 0; t < npt; ++t) {
        int kvb = t * 32;
        __syncthreads();
        {   // stage V^T cooperatively: [64 d][32 kv]
            int kv = tid >> 3, ds = tid & 7;
            HalfPack p; p.u = *(const uint4*)(vbase + (size_t)(kvb + kv) * ts + ds * 8);
#pragma unroll
            for (int j = 0; j < 8; ++j) sVT[(ds * 8 + j) * 32 + kv] = p.h[j];
        }
        __syncthreads();

        // scores: two 16-col subtiles, K-fragments straight from global
        v8f s0 = {}, s1 = {};
#pragma unroll
        for (int st = 0; st < 2; ++st) {
            v8f sc = {};
            int krow = kvb + st * 16 + lo;
#pragma unroll
            for (int ks = 0; ks < 2; ++ks) {
                const __bf16* kp = kbase + (size_t)krow * ts + ks * 32 + hi * 16;
                FragU f; f.u[0] = *(const uint4*)kp; f.u[1] = *(const uint4*)(kp + 8);
                sc = __builtin_amdgcn_wmma_f32_16x16x32_bf16(
                    false, qa[ks], false, f.v, (short)0, sc, false, false);
            }
            if (st == 0) s0 = sc; else s1 = sc;
        }

        // online softmax over the 32 new columns
#pragma unroll
        for (int r = 0; r < 8; ++r) {
            int row = qb + w * 16 + r + 8 * hi;
            float a = s0[r] * 0.125f;            // hd^-0.5 = 1/8
            float c = s1[r] * 0.125f;
            if (kvb + lo > row)      a = -1e30f;
            if (kvb + 16 + lo > row) c = -1e30f;
            float loc = fmaxf(a, c);
            loc = fmaxf(loc, __shfl_xor(loc, 1, 16));
            loc = fmaxf(loc, __shfl_xor(loc, 2, 16));
            loc = fmaxf(loc, __shfl_xor(loc, 4, 16));
            loc = fmaxf(loc, __shfl_xor(loc, 8, 16));
            float mn = fmaxf(m[r], loc);
            float fs = __expf(m[r] - mn);
            float p0 = __expf(a - mn);
            float p1 = __expf(c - mn);
            float sum = p0 + p1;
            sum += __shfl_xor(sum, 1, 16);
            sum += __shfl_xor(sum, 2, 16);
            sum += __shfl_xor(sum, 4, 16);
            sum += __shfl_xor(sum, 8, 16);
            l[r] = l[r] * fs + sum;
            m[r] = mn;
            o[0][r] *= fs; o[1][r] *= fs; o[2][r] *= fs; o[3][r] *= fs;
            pb[(r + 8 * hi) * 32 + lo]      = (__bf16)p0;
            pb[(r + 8 * hi) * 32 + 16 + lo] = (__bf16)p1;
        }

        // P (16x32) @ V (32x64): A-frag from pbuf, B-frags from sVT
        FragU fp;
        {
            int base = lo * 32 + hi * 16;
            fp.u[0] = *(const uint4*)(pb + base);
            fp.u[1] = *(const uint4*)(pb + base + 8);
        }
#pragma unroll
        for (int d = 0; d < 4; ++d) {
            int base = (d * 16 + lo) * 32 + hi * 16;
            FragU fv; fv.u[0] = *(const uint4*)(sVT + base); fv.u[1] = *(const uint4*)(sVT + base + 8);
            o[d] = __builtin_amdgcn_wmma_f32_16x16x32_bf16(
                false, fp.v, false, fv.v, (short)0, o[d], false, false);
        }
    }

    // normalize + write bf16 [8192][1024]
#pragma unroll
    for (int d = 0; d < 4; ++d)
#pragma unroll
        for (int r = 0; r < 8; ++r) {
            int row = qb + w * 16 + r + 8 * hi;
            float v = o[d][r] / l[r];
            attn[((size_t)b * 2048 + row) * 1024 + h * 64 + d * 16 + lo] = (__bf16)v;
        }
}

// ---------------------------------------------------------------------------
// Router: logits = y @ w_router + b_router (E=8), softmax -> rw[8192][8]
// ---------------------------------------------------------------------------
__global__ __launch_bounds__(256) void router_kernel(const __bf16* __restrict__ y,
                                                     const float* __restrict__ wr,
                                                     const float* __restrict__ br,
                                                     float* __restrict__ rw)
{
    __shared__ float red[256 * 8];
    __shared__ float probs[8];
    int t = threadIdx.x;
    size_t tok = blockIdx.x;
    float acc[8] = {};
#pragma unroll
    for (int i = 0; i < 4; ++i) {
        int hh = t + i * 256;
        float yv = (float)y[tok * 1024 + hh];
        const float* wrow = wr + (size_t)hh * 8;
#pragma unroll
        for (int e = 0; e < 8; ++e) acc[e] += yv * wrow[e];
    }
#pragma unroll
    for (int e = 0; e < 8; ++e) red[t * 8 + e] = acc[e];
    __syncthreads();
    for (int o = 128; o > 0; o >>= 1) {
        if (t < o)
#pragma unroll
            for (int e = 0; e < 8; ++e) red[t * 8 + e] += red[(t + o) * 8 + e];
        __syncthreads();
    }
    if (t == 0) {
        float lg[8], mx = -1e30f, sm = 0.f;
#pragma unroll
        for (int e = 0; e < 8; ++e) { lg[e] = red[e] + br[e]; mx = fmaxf(mx, lg[e]); }
#pragma unroll
        for (int e = 0; e < 8; ++e) { lg[e] = __expf(lg[e] - mx); sm += lg[e]; }
#pragma unroll
        for (int e = 0; e < 8; ++e) probs[e] = lg[e] / sm;
    }
    __syncthreads();
    if (t < 8) rw[tok * 8 + t] = probs[t];
}

// ---------------------------------------------------------------------------
// Final: al = al1 @ w_al2 + b_al2 ; gate ; LayerNorm -> f32 out
// ---------------------------------------------------------------------------
__global__ __launch_bounds__(256) void final_kernel(const float* __restrict__ al1,
                                                    const float* __restrict__ wal2,
                                                    const float* __restrict__ bal2,
                                                    const float* __restrict__ h2,
                                                    const float* __restrict__ g,
                                                    const float* __restrict__ bb,
                                                    float* __restrict__ out)
{
    __shared__ float red[256];
    int t = threadIdx.x;
    size_t tok = blockIdx.x;
    red[t] = al1[tok * 256 + t] * wal2[t];
    __syncthreads();
    for (int o = 128; o > 0; o >>= 1) { if (t < o) red[t] += red[t + o]; __syncthreads(); }
    float al = red[0] + bal2[0];
    float safe = (al > 0.8f) ? 1.f : 0.f;
    __syncthreads();

    float v[4], s = 0.f;
#pragma unroll
    for (int i = 0; i < 4; ++i) { v[i] = h2[tok * 1024 + t + i * 256] * safe; s += v[i]; }
    red[t] = s; __syncthreads();
    for (int o = 128; o > 0; o >>= 1) { if (t < o) red[t] += red[t + o]; __syncthreads(); }
    float mu = red[0] * (1.f / 1024.f);
    __syncthreads();
    float s2 = 0.f;
#pragma unroll
    for (int i = 0; i < 4; ++i) { float d = v[i] - mu; s2 += d * d; }
    red[t] = s2; __syncthreads();
    for (int o = 128; o > 0; o >>= 1) { if (t < o) red[t] += red[t + o]; __syncthreads(); }
    float inv = rsqrtf(red[0] * (1.f / 1024.f) + 1e-12f);
#pragma unroll
    for (int i = 0; i < 4; ++i) {
        int c = t + i * 256;
        out[tok * 1024 + c] = (v[i] - mu) * inv * g[c] + bb[c];
    }
}

// ---------------------------------------------------------------------------
// Host launcher
// ---------------------------------------------------------------------------
extern "C" void kernel_launch(void* const* d_in, const int* in_sizes, int n_in,
                              void* d_out, int out_size, void* d_ws, size_t ws_size,
                              hipStream_t stream)
{
    (void)in_sizes; (void)n_in; (void)out_size; (void)ws_size;
    const float* inputs   = (const float*)d_in[0];
    const float* ln1_g    = (const float*)d_in[1];
    const float* ln1_b    = (const float*)d_in[2];
    const float* w_qkv    = (const float*)d_in[3];
    const float* b_qkv    = (const float*)d_in[4];
    const float* w_out    = (const float*)d_in[5];
    const float* b_out    = (const float*)d_in[6];
    const float* ln2_g    = (const float*)d_in[7];
    const float* ln2_b    = (const float*)d_in[8];
    const float* w_router = (const float*)d_in[9];
    const float* b_router = (const float*)d_in[10];
    const float* w_exp    = (const float*)d_in[11];
    const float* b_exp    = (const float*)d_in[12];
    const float* w_al1    = (const float*)d_in[13];
    const float* b_al1    = (const float*)d_in[14];
    const float* w_al2    = (const float*)d_in[15];
    const float* b_al2    = (const float*)d_in[16];
    const float* lnf_g    = (const float*)d_in[17];
    const float* lnf_b    = (const float*)d_in[18];
    float* out = (float*)d_out;

    const size_t T = 8192, H = 1024;
    char* ws = (char*)d_ws;
    size_t off = 0;
    auto alloc = [&](size_t bytes) -> char* {
        char* p = ws + off; off += (bytes + 255) & ~(size_t)255; return p;
    };
    __bf16* x_bf    = (__bf16*)alloc(T * H * 2);
    __bf16* qkv_bf  = (__bf16*)alloc(T * 3 * H * 2);
    __bf16* attn_bf = (__bf16*)alloc(T * H * 2);
    float*  h_f     = (float*) alloc(T * H * 4);
    __bf16* y_bf    = (__bf16*)alloc(T * H * 2);
    float*  rw_f    = (float*) alloc(T * 8 * 4);
    float*  h2_f    = (float*) alloc(T * H * 4);
    __bf16* h2_bf   = (__bf16*)alloc(T * H * 2);
    float*  al1_f   = (float*) alloc(T * 256 * 4);
    __bf16* wq_t    = (__bf16*)alloc(H * 3 * H * 2);   // [3072][1024]
    __bf16* wo_t    = (__bf16*)alloc(H * H * 2);       // [1024][1024]
    __bf16* we_t    = (__bf16*)alloc(8 * H * H * 2);   // 8 x [1024][1024]
    __bf16* wa1_t   = (__bf16*)alloc(H * 256 * 2);     // [256][1024]

    // weight conversions (transpose to [N][K] bf16; stays resident in L2)
    cvt_transpose_kernel<<<(int)((H * 3 * H) / 256), 256, 0, stream>>>(w_qkv, wq_t, 1024, 3072);
    cvt_transpose_kernel<<<(int)((H * H) / 256), 256, 0, stream>>>(w_out, wo_t, 1024, 1024);
    for (int e = 0; e < 8; ++e)
        cvt_transpose_kernel<<<(int)((H * H) / 256), 256, 0, stream>>>(
            w_exp + (size_t)e * H * H, we_t + (size_t)e * H * H, 1024, 1024);
    cvt_transpose_kernel<<<(int)((H * 256) / 256), 256, 0, stream>>>(w_al1, wa1_t, 1024, 256);

    auto gemm = [&](const __bf16* A, int lda, const __bf16* BT, int ldbt,
                    float* Cf, __bf16* Cb, int ldc,
                    const float* bias, const float* resid,
                    const float* rs, int rss, int M, int N, int K, int flags) {
        dim3 grid(N / 128, M / 128);
        gemm_bf16_kernel<<<grid, 256, 0, stream>>>(A, lda, BT, ldbt, Cf, Cb, ldc,
                                                   bias, resid, rs, rss, M, N, K, flags);
    };

    // 1) x = LN1(inputs) -> bf16
    ln_bf16_kernel<<<(int)T, 256, 0, stream>>>(inputs, ln1_g, ln1_b, x_bf);
    // 2) qkv = x @ w_qkv + b_qkv -> bf16 [8192][3072]
    gemm(x_bf, 1024, wq_t, 1024, nullptr, qkv_bf, 3072,
         b_qkv, nullptr, nullptr, 0, 8192, 3072, 1024, EPI_BIAS);
    // 3) causal flash attention -> attn bf16 [8192][1024]
    attn_kernel<<<dim3(16, 16, 4), 256, 0, stream>>>(qkv_bf, attn_bf);
    // 4) h = inputs + attn @ w_out + b_out -> f32
    gemm(attn_bf, 1024, wo_t, 1024, h_f, nullptr, 1024,
         b_out, inputs, nullptr, 0, 8192, 1024, 1024, EPI_BIAS | EPI_RESID);
    // 5) y = LN2(h) -> bf16
    ln_bf16_kernel<<<(int)T, 256, 0, stream>>>(h_f, ln2_g, ln2_b, y_bf);
    // 6) router softmax
    router_kernel<<<(int)T, 256, 0, stream>>>(y_bf, w_router, b_router, rw_f);
    // 7) h2 = h + sum_e rw_e * (y @ w_exp[e] + b_exp[e])
    for (int e = 0; e < 8; ++e) {
        int flags = EPI_BIAS | EPI_ROWSCALE | (e == 0 ? EPI_RESID : EPI_ACCUM);
        gemm(y_bf, 1024, we_t + (size_t)e * H * H, 1024,
             h2_f, (e == 7) ? h2_bf : nullptr, 1024,
             b_exp + (size_t)e * H, h_f, rw_f + e, 8,
             8192, 1024, 1024, flags);
    }
    // 8) al1 = gelu(h2 @ w_al1 + b_al1) -> f32 [8192][256]
    gemm(h2_bf, 1024, wa1_t, 1024, al1_f, nullptr, 256,
         b_al1, nullptr, nullptr, 0, 8192, 256, 1024, EPI_BIAS | EPI_GELU);
    // 9) gate + final LN -> out f32
    final_kernel<<<(int)T, 256, 0, stream>>>(al1_f, w_al2, b_al2, h2_f, lnf_g, lnf_b, out);
}
